// Head_69930657513851
// MI455X (gfx1250) — compile-verified, matrix operand order checked by reading, not run on hardware
//
#include <hip/hip_runtime.h>
#include <math.h>

typedef __attribute__((ext_vector_type(2)))  __fp16   v2hf;  // cvt_pkrtz result type
typedef __attribute__((ext_vector_type(4)))  _Float16 v4h;
typedef __attribute__((ext_vector_type(8)))  _Float16 v8h;
typedef __attribute__((ext_vector_type(16))) _Float16 v16h;
typedef __attribute__((ext_vector_type(8)))  float    v8f;

#define BATCH 4
#define SEQ   4096
#define DHEAD 128
#define KST   136   // K tile LDS stride (halfs): 272B rows -> conflict-free b128 frag loads
#define VST   40    // V^T tile LDS stride (halfs): 80B rows
#define PST   40    // P staging stride (halfs)
#define KBUF  (32 * KST)
#define VBUF  (DHEAD * VST)

// VALU-pipe xor-shuffle within 16-lane rows (DPP ROW_XMASK), replaces ds_bpermute.
#define DPP_XORF(v, m)                                                        \
    __int_as_float(__builtin_amdgcn_update_dpp(                               \
        0, __float_as_int(v), 0x160 | (m), 0xf, 0xf, true))

__device__ __forceinline__ float rowmax16(float x) {
    x = fmaxf(x, DPP_XORF(x, 1));
    x = fmaxf(x, DPP_XORF(x, 2));
    x = fmaxf(x, DPP_XORF(x, 4));
    x = fmaxf(x, DPP_XORF(x, 8));
    return x;
}
__device__ __forceinline__ float rowsum16(float x) {
    x += DPP_XORF(x, 1);
    x += DPP_XORF(x, 2);
    x += DPP_XORF(x, 4);
    x += DPP_XORF(x, 8);
    return x;
}

// Load a 16-half A/B fragment from LDS: two contiguous 16B runs at p and p+16 halfs.
__device__ __forceinline__ v16h ld_frag16(const _Float16* p) {
    v8h lo = *(const v8h*)(p);
    v8h hi = *(const v8h*)(p + 16);
    v16h r;
#pragma unroll
    for (int i = 0; i < 8; ++i) { r[i] = lo[i]; r[i + 8] = hi[i]; }
    return r;
}

__global__ __launch_bounds__(128)
void fa_causal_kernel(const float* __restrict__ Kg,
                      const float* __restrict__ Qg,
                      const float* __restrict__ Vg,
                      float* __restrict__ Og) {
    __shared__ __align__(16) _Float16 sK[2 * KBUF];     // double-buffered, row-major [key][d]
    __shared__ __align__(16) _Float16 sV[2 * VBUF];     // double-buffered, transposed [d][key]
    __shared__ __align__(16) _Float16 sP[4 * 16 * PST]; // per-wave P staging

    const int tid  = threadIdx.x;
    const int lane = tid & 31;
    const int w    = tid >> 5;
    const int hi   = (lane >> 4) & 1;
    const int col  = lane & 15;
    const int base = blockIdx.x * 64;       // Q-block first row
    const int b    = blockIdx.y;
    const int q0   = base + w * 16;         // this wave's first Q row
    const float scale = 0.08838834764831844f; // 1/sqrt(128)

    const float* kbase = Kg + (size_t)b * SEQ * DHEAD;
    const float* vbase = Vg + (size_t)b * SEQ * DHEAD;
    _Float16* myP = sP + w * 16 * PST;

    // ---- Q tile as four 16x32 f16 A-fragments (scale folded in, pk converts) ----
    const float* qrow = Qg + ((size_t)b * SEQ + (size_t)(q0 + col)) * DHEAD;
    v16h aq[4];
#pragma unroll
    for (int c = 0; c < 4; ++c) {
        const int d0 = c * 32 + hi * 8;
        float4 f0 = *(const float4*)(qrow + d0);
        float4 f1 = *(const float4*)(qrow + d0 + 4);
        float4 f2 = *(const float4*)(qrow + d0 + 16);
        float4 f3 = *(const float4*)(qrow + d0 + 20);
        v2hf p0 = __builtin_amdgcn_cvt_pkrtz(f0.x * scale, f0.y * scale);
        v2hf p1 = __builtin_amdgcn_cvt_pkrtz(f0.z * scale, f0.w * scale);
        v2hf p2 = __builtin_amdgcn_cvt_pkrtz(f1.x * scale, f1.y * scale);
        v2hf p3 = __builtin_amdgcn_cvt_pkrtz(f1.z * scale, f1.w * scale);
        v2hf p4 = __builtin_amdgcn_cvt_pkrtz(f2.x * scale, f2.y * scale);
        v2hf p5 = __builtin_amdgcn_cvt_pkrtz(f2.z * scale, f2.w * scale);
        v2hf p6 = __builtin_amdgcn_cvt_pkrtz(f3.x * scale, f3.y * scale);
        v2hf p7 = __builtin_amdgcn_cvt_pkrtz(f3.z * scale, f3.w * scale);
        v16h a;
        a[0]  = (_Float16)p0[0]; a[1]  = (_Float16)p0[1];
        a[2]  = (_Float16)p1[0]; a[3]  = (_Float16)p1[1];
        a[4]  = (_Float16)p2[0]; a[5]  = (_Float16)p2[1];
        a[6]  = (_Float16)p3[0]; a[7]  = (_Float16)p3[1];
        a[8]  = (_Float16)p4[0]; a[9]  = (_Float16)p4[1];
        a[10] = (_Float16)p5[0]; a[11] = (_Float16)p5[1];
        a[12] = (_Float16)p6[0]; a[13] = (_Float16)p6[1];
        a[14] = (_Float16)p7[0]; a[15] = (_Float16)p7[1];
        aq[c] = a;
    }

    // ---- accumulators / softmax state ----
    v8f oacc[8];
#pragma unroll
    for (int c = 0; c < 8; ++c)
#pragma unroll
        for (int r = 0; r < 8; ++r) oacc[c][r] = 0.0f;
    float m_i[8], l_i[8];
#pragma unroll
    for (int r = 0; r < 8; ++r) { m_i[r] = -INFINITY; l_i[r] = 0.0f; }

    // Prefetch registers for the double-buffered pipeline (16 float4 = 64 VGPRs).
    float4 kr[8], vr[8];

    auto load_tile_regs = [&](int j) {
#pragma unroll
        for (int i = 0; i < 8; ++i) {
            const int f   = tid + i * 128;
            const int row = f >> 5;
            const int d   = (f & 31) * 4;
            kr[i] = *(const float4*)(kbase + (size_t)(j + row) * DHEAD + d);
            vr[i] = *(const float4*)(vbase + (size_t)(j + row) * DHEAD + d);
        }
    };
    auto store_tile = [&](int buf) {
        _Float16* dK = sK + buf * KBUF;
        _Float16* dV = sV + buf * VBUF;
#pragma unroll
        for (int i = 0; i < 8; ++i) {
            const int f   = tid + i * 128;
            const int row = f >> 5;
            const int d   = (f & 31) * 4;
            v2hf ka = __builtin_amdgcn_cvt_pkrtz(kr[i].x, kr[i].y);
            v2hf kb = __builtin_amdgcn_cvt_pkrtz(kr[i].z, kr[i].w);
            v4h hk;
            hk[0] = (_Float16)ka[0]; hk[1] = (_Float16)ka[1];
            hk[2] = (_Float16)kb[0]; hk[3] = (_Float16)kb[1];
            *(v4h*)(&dK[row * KST + d]) = hk;
            v2hf va = __builtin_amdgcn_cvt_pkrtz(vr[i].x, vr[i].y);
            v2hf vb = __builtin_amdgcn_cvt_pkrtz(vr[i].z, vr[i].w);
            dV[(d + 0) * VST + row] = (_Float16)va[0];
            dV[(d + 1) * VST + row] = (_Float16)va[1];
            dV[(d + 2) * VST + row] = (_Float16)vb[0];
            dV[(d + 3) * VST + row] = (_Float16)vb[1];
        }
    };

    const int jend = base + 64;

    // prologue: stage tile j=0
    load_tile_regs(0);
    store_tile(0);
    __syncthreads();

    int cur = 0;
    for (int j = 0; j < jend; j += 32) {
        const bool pf = (j + 32 < jend);
        if (pf) load_tile_regs(j + 32);   // global loads in flight during compute

        if (j <= q0 + 15) {               // wave-uniform causal skip
            const _Float16* cK = sK + cur * KBUF;
            const _Float16* cV = sV + cur * VBUF;

            // S = Q * K^T : two 16x16 tiles over 4 d-chunks, with B-fragment
            // double buffering so ds_loads stay in flight across each WMMA pair.
            v8f s0, s1;
#pragma unroll
            for (int r = 0; r < 8; ++r) { s0[r] = 0.0f; s1[r] = 0.0f; }
            v16h bk0 = ld_frag16(&cK[col * KST + hi * 8]);
            v16h bk1 = ld_frag16(&cK[(16 + col) * KST + hi * 8]);
#pragma unroll
            for (int c = 0; c < 4; ++c) {
                v16h nk0, nk1;
                if (c < 3) {
                    nk0 = ld_frag16(&cK[col * KST + (c + 1) * 32 + hi * 8]);
                    nk1 = ld_frag16(&cK[(16 + col) * KST + (c + 1) * 32 + hi * 8]);
                }
                s0 = __builtin_amdgcn_wmma_f32_16x16x32_f16(false, aq[c], false, bk0,
                                                            (short)0, s0, false, false);
                s1 = __builtin_amdgcn_wmma_f32_16x16x32_f16(false, aq[c], false, bk1,
                                                            (short)0, s1, false, false);
                bk0 = nk0; bk1 = nk1;
            }

            // causal mask on diagonal blocks only
            if (j + 31 > q0) {
#pragma unroll
                for (int r = 0; r < 8; ++r) {
                    const int q = q0 + r + hi * 8;
                    if (j + col > q)      s0[r] = -INFINITY;
                    if (j + 16 + col > q) s1[r] = -INFINITY;
                }
            }

            // online softmax; row stats via DPP ROW_XMASK reductions (VALU pipe)
            float alpha[8];
#pragma unroll
            for (int r = 0; r < 8; ++r) {
                float rm = rowmax16(fmaxf(s0[r], s1[r]));
                const float mn = fmaxf(m_i[r], rm);
                alpha[r] = __expf(m_i[r] - mn);
                m_i[r] = mn;
                s0[r] = __expf(s0[r] - mn);
                s1[r] = __expf(s1[r] - mn);
                l_i[r] = l_i[r] * alpha[r] + rowsum16(s0[r] + s1[r]);
            }

            // stage P (f16) through per-wave LDS to re-layout C/D -> A fragment
#pragma unroll
            for (int r = 0; r < 8; ++r) {
                const int row = r + hi * 8;
                v2hf ph = __builtin_amdgcn_cvt_pkrtz(s0[r], s1[r]);
                myP[row * PST + col]      = (_Float16)ph[0];
                myP[row * PST + 16 + col] = (_Float16)ph[1];
            }
            v16h ap = ld_frag16(&myP[col * PST + hi * 8]);

            // O = alpha*O + P * V, V-fragment double buffered; the alpha
            // v_pk_mul rescales fill the ds_load shadow of the next chunk.
            v16h bv = ld_frag16(&cV[col * VST + hi * 8]);
#pragma unroll
            for (int c = 0; c < 8; ++c) {
                v16h nbv;
                if (c < 7)
                    nbv = ld_frag16(&cV[((c + 1) * 16 + col) * VST + hi * 8]);
                v8f o = oacc[c];
#pragma unroll
                for (int r = 0; r < 8; ++r) o[r] *= alpha[r];
                oacc[c] = __builtin_amdgcn_wmma_f32_16x16x32_f16(false, ap, false, bv,
                                                                 (short)0, o, false, false);
                bv = nbv;
            }
        }

        if (pf) store_tile(cur ^ 1);      // convert + stage next tile after compute
        __syncthreads();                  // single barrier per iteration
        cur ^= 1;
    }

    // ---- epilogue: normalize and store f32 ----
    float* orow = Og + ((size_t)b * SEQ + (size_t)q0) * DHEAD;
#pragma unroll
    for (int r = 0; r < 8; ++r) {
        const float inv = 1.0f / l_i[r];
        const int row = r + hi * 8;
#pragma unroll
        for (int c = 0; c < 8; ++c)
            orow[(size_t)row * DHEAD + c * 16 + col] = oacc[c][r] * inv;
    }
}

extern "C" void kernel_launch(void* const* d_in, const int* in_sizes, int n_in,
                              void* d_out, int out_size, void* d_ws, size_t ws_size,
                              hipStream_t stream) {
    (void)in_sizes; (void)n_in; (void)d_ws; (void)ws_size; (void)out_size;
    const float* key   = (const float*)d_in[0];
    const float* query = (const float*)d_in[1];
    const float* value = (const float*)d_in[2];
    float* out = (float*)d_out;
    dim3 grid(SEQ / 64, BATCH);
    dim3 block(128);
    fa_causal_kernel<<<grid, block, 0, stream>>>(key, query, value, out);
}
